// LSTM_51144470560975
// MI455X (gfx1250) — compile-verified, matrix operand order checked by reading, not run on hardware
//
#include <hip/hip_runtime.h>
#include <hip/hip_bf16.h>

// ---------------------------------------------------------------------------
// Types for CDNA5 WMMA (wave32): v_wmma_f32_16x16x32_bf16
// ---------------------------------------------------------------------------
typedef __bf16 bhalf;
typedef __attribute__((ext_vector_type(16))) __bf16 v16bf;
typedef __attribute__((ext_vector_type(8)))  float  v8f;
typedef __attribute__((ext_vector_type(4)))  float  f32x4;
typedef __attribute__((ext_vector_type(4)))  unsigned int u32x4;

union FragU { v16bf bf; u32x4 q[2]; };
union AccU  { v8f v; f32x4 h[2]; };

__device__ __forceinline__ bhalf f2bf(float x) {
    unsigned u = __float_as_uint(x);
    unsigned r = (u + 0x7FFFu + ((u >> 16) & 1u)) >> 16;   // round-to-nearest-even
    unsigned short s = (unsigned short)r;
    return __builtin_bit_cast(bhalf, s);
}

#if __has_builtin(__builtin_amdgcn_tanhf)
#define HAVE_HW_TANH 1
#else
#define HAVE_HW_TANH 0
#endif

__device__ __forceinline__ float fast_tanh(float x) {
#if HAVE_HW_TANH
    return __builtin_amdgcn_tanhf(x);        // V_TANH_F32 on gfx1250
#else
    return tanhf(x);
#endif
}

// sigmoid(x) = 0.5*tanh(x/2) + 0.5  -> mul + v_tanh_f32 + fma (no divide)
__device__ __forceinline__ float fast_sig(float x) {
#if HAVE_HW_TANH
    return __builtin_fmaf(0.5f, __builtin_amdgcn_tanhf(0.5f * x), 0.5f);
#elif __has_builtin(__builtin_amdgcn_rcpf)
    return __builtin_amdgcn_rcpf(1.f + __expf(-x));
#else
    return 1.f / (1.f + __expf(-x));
#endif
}

// 16-bit A-matrix 16x32 layout (ISA 7.12.2): lane L (L<16) holds row L,
// K = {0..7} and {16..23}; lane L>=16 holds row L-16, K = {8..15},{24..31}.
// Loading B^T (row-major [N][K]) with the same pattern gives the B fragment.
__device__ __forceinline__ v16bf load_frag(const bhalf* base, int ld,
                                           int row0, int kbase, int lane) {
    const bhalf* p = base + (size_t)(row0 + (lane & 15)) * ld
                          + kbase + ((lane >> 4) & 1) * 8;
    FragU f;
    f.q[0] = *(const u32x4*)(p);
    f.q[1] = *(const u32x4*)(p + 16);
    return f.bf;
}

__device__ __forceinline__ v8f wmma_bf16(v16bf a, v16bf b, v8f c) {
    return __builtin_amdgcn_wmma_f32_16x16x32_bf16(
        /*neg_a=*/false, a, /*neg_b=*/false, b,
        /*c_mod=*/(short)0, c, /*reuse_a=*/false, /*reuse_b=*/false);
}

#define HH 128   // hidden
#define GG 512   // 4*H gates

// Swizzled xg layout (WMMA C-fragment order, 16x16 tiles):
//   xg[dir][tileRow][g in 0..511][r in 0..15],  tileRow = t*(Bn/16) + btile
// so lane (col g, rows Mv..Mv+7) reads/writes 8 contiguous floats.

// ---------------------------------------------------------------------------
// f32 -> bf16 weight conversion
// ---------------------------------------------------------------------------
__global__ void convert_kernel(const float* __restrict__ s, bhalf* __restrict__ d, int n) {
    int i = blockIdx.x * blockDim.x + threadIdx.x;
    if (i < n) d[i] = f2bf(s[i]);
}

// ---------------------------------------------------------------------------
// Layer-0 input projection into swizzled xg.
// One block = (t, 16-batch tile); thread g computes its gate column for all
// 16 rows (x tile staged in LDS) and writes 64 contiguous bytes.
// ---------------------------------------------------------------------------
__global__ __launch_bounds__(512) void xgate0_kernel(
    const float* __restrict__ x,
    const float* __restrict__ wf, const float* __restrict__ bif, const float* __restrict__ bhf,
    const float* __restrict__ wr, const float* __restrict__ bir, const float* __restrict__ bhr,
    float* __restrict__ xg, int T, int Bn, int Cc)
{
    const int dir = blockIdx.y;
    const int tr  = blockIdx.x;              // tileRow
    const int nbt = Bn >> 4;
    const int t   = tr / nbt;
    const int b0  = (tr % nbt) * 16;
    const int tid = threadIdx.x;

    __shared__ float xs[16][12];             // 16 rows x 9 inputs (padded)
    if (tid < 16 * 9) {
        int br = tid / 9, i = tid % 9;
        xs[br][i] = x[((size_t)(b0 + br) * Cc + i) * T + t];
    }
    __syncthreads();

    const int g = tid;
    const float* w = dir ? wr : wf;
    float wreg[9];
#pragma unroll
    for (int i = 0; i < 9; ++i) wreg[i] = w[g * Cc + i];
    const float bias = dir ? (bir[g] + bhr[g]) : (bif[g] + bhf[g]);

    float vals[16];
#pragma unroll
    for (int br = 0; br < 16; ++br) {
        float acc = bias;
#pragma unroll
        for (int i = 0; i < 9; ++i) acc += xs[br][i] * wreg[i];
        vals[br] = acc;
    }

    const size_t NT = (size_t)T * nbt;
    float* dst = xg + (((size_t)dir * NT + tr) * GG + g) * 16;
#pragma unroll
    for (int q = 0; q < 4; ++q) {
        f32x4 v4 = { vals[q * 4 + 0], vals[q * 4 + 1], vals[q * 4 + 2], vals[q * 4 + 3] };
        *(f32x4*)(dst + q * 4) = v4;
    }
}

// ---------------------------------------------------------------------------
// Serial LSTM scan (both directions concurrently via blockIdx.y).
// One WG = 16 batch rows, 16 waves. Recurrent bf16 weights live in VGPRs
// (8 fragments/wave), h state lives in LDS as bf16; per step each wave does
// 8 v_wmma_f32_16x16x32_bf16 + 4 b128 acc-init loads (swizzled xg).
// Gate math uses V_TANH_F32 only (no divides) on the serial critical path.
// ---------------------------------------------------------------------------
__global__ __launch_bounds__(512) void lstm_scan_kernel(
    const float* __restrict__ xg,    // swizzled [2][T*Bn/16][512][16] f32
    const bhalf* __restrict__ whh,   // [2][512][128] bf16
    bhalf* __restrict__ out0,        // [T][B][256] bf16 or nullptr
    float* __restrict__ hfinal,      // [B][256] f32 or nullptr
    int T, int Bn)
{
    __shared__ __align__(16) bhalf h_lds[16 * HH];   // 4 KB
    __shared__ float c_lds[16 * HH];                 // 8 KB
    __shared__ float g_lds[16 * GG];                 // 32 KB

    const int tid  = threadIdx.x;
    const int lane = tid & 31;
    const int wave = tid >> 5;
    const int dir  = blockIdx.y;
    const int btile = blockIdx.x;
    const int b0   = btile * 16;
    const int nbt  = Bn >> 4;

    for (int k = tid; k < 16 * HH; k += 512) { h_lds[k] = f2bf(0.f); c_lds[k] = 0.f; }
    __syncthreads();

    const bhalf* whh_d = whh + (size_t)dir * GG * HH;
    const int nt0 = wave * 2;           // 16 waves x 2 n-tiles = 32 tiles = 512 cols

    // Resident recurrent-weight fragments: 2 n-tiles x 4 k-chunks (K=128)
    v16bf bfrag[2][4];
#pragma unroll
    for (int nt = 0; nt < 2; ++nt)
#pragma unroll
        for (int kc = 0; kc < 4; ++kc)
            bfrag[nt][kc] = load_frag(whh_d, HH, (nt0 + nt) * 16, kc * 32, lane);

    const int Mv    = (lane >> 4) * 8;   // C/D layout: M = v + 8*(lane>=16)
    const int ncolL = lane & 15;

    for (int step = 0; step < T; ++step) {
        const int t = dir ? (T - 1 - step) : step;
        const float* xgt = xg + (((size_t)dir * T + t) * nbt + btile) * (GG * 16);

        // accumulator initialized with precomputed input gates (swizzled, b128 loads)
        v8f acc[2];
#pragma unroll
        for (int nt = 0; nt < 2; ++nt) {
            const float* p = xgt + (size_t)((nt0 + nt) * 16 + ncolL) * 16 + Mv;
            AccU au;
            au.h[0] = *(const f32x4*)(p);
            au.h[1] = *(const f32x4*)(p + 4);
            acc[nt] = au.v;
        }

        // A fragments = h tile (16x128 bf16) from LDS
        v16bf afrag[4];
#pragma unroll
        for (int kc = 0; kc < 4; ++kc)
            afrag[kc] = load_frag(h_lds, HH, 0, kc * 32, lane);

#pragma unroll
        for (int nt = 0; nt < 2; ++nt)
#pragma unroll
            for (int kc = 0; kc < 4; ++kc)
                acc[nt] = wmma_bf16(afrag[kc], bfrag[nt][kc], acc[nt]);

        // publish raw gates
#pragma unroll
        for (int nt = 0; nt < 2; ++nt) {
            const int ncol = (nt0 + nt) * 16 + ncolL;
#pragma unroll
            for (int v = 0; v < 8; ++v)
                g_lds[(Mv + v) * GG + ncol] = acc[nt][v];
        }
        __syncthreads();

        // elementwise cell update (2048 elems / 512 threads)
#pragma unroll
        for (int k = 0; k < 4; ++k) {
            const int id  = tid + k * 512;
            const int row = id >> 7;
            const int col = id & (HH - 1);
            const float i_ = fast_sig(g_lds[row * GG + col]);
            const float f_ = fast_sig(g_lds[row * GG + HH + col]);
            const float g_ = fast_tanh(g_lds[row * GG + 2 * HH + col]);
            const float o_ = fast_sig(g_lds[row * GG + 3 * HH + col]);
            const float c_ = f_ * c_lds[row * HH + col] + i_ * g_;
            c_lds[row * HH + col] = c_;
            const float h_ = o_ * fast_tanh(c_);
            h_lds[row * HH + col] = f2bf(h_);
            if (out0)
                out0[((size_t)t * Bn + b0 + row) * 256 + dir * HH + col] = f2bf(h_);
            if (hfinal && step == T - 1)
                hfinal[(size_t)(b0 + row) * 256 + dir * HH + col] = h_;
        }
        __syncthreads();
    }
}

// ---------------------------------------------------------------------------
// Layer-1 input projection: xg1[dir] = out0 @ w_ih_l1(dir)^T + bias
// M = T*B = 524288, N = 512, K = 256. One wave per 16x16 tile strip
// (4 M-tiles reuse the 8 resident B fragments). 32 WMMA per wave.
// Output stored directly in swizzled fragment order (2 b128 stores / tile).
// ---------------------------------------------------------------------------
__global__ __launch_bounds__(256) void xgate1_kernel(
    const bhalf* __restrict__ act,   // out0 [M][256] bf16
    const bhalf* __restrict__ wih,   // [2][512][256] bf16
    const float* __restrict__ bif, const float* __restrict__ bhf,
    const float* __restrict__ bir, const float* __restrict__ bhr,
    float* __restrict__ xg, int Mrows)
{
    const int lane = threadIdx.x & 31;
    const int wave = threadIdx.x >> 5;
    const int dir  = blockIdx.y;
    const int gid  = blockIdx.x * 8 + wave;
    const int mgroup = gid >> 5;           // 64 rows each
    const int ntile  = gid & 31;

    const bhalf* wd = wih + (size_t)dir * GG * 256;
    v16bf bfrag[8];
#pragma unroll
    for (int kc = 0; kc < 8; ++kc)
        bfrag[kc] = load_frag(wd, 256, ntile * 16, kc * 32, lane);

    const int ncol = ntile * 16 + (lane & 15);
    const float bias = dir ? (bir[ncol] + bhr[ncol]) : (bif[ncol] + bhf[ncol]);
    const int Mv = (lane >> 4) * 8;
    const size_t NT = (size_t)Mrows >> 4;

#pragma unroll
    for (int mi = 0; mi < 4; ++mi) {
        const int m = mgroup * 64 + mi * 16;
        v8f acc;
#pragma unroll
        for (int v = 0; v < 8; ++v) acc[v] = bias;
#pragma unroll
        for (int kc = 0; kc < 8; ++kc) {
            v16bf a = load_frag(act, 256, m, kc * 32, lane);
            acc = wmma_bf16(a, bfrag[kc], acc);
        }
        // swizzled store: tileRow = m/16, 8 contiguous floats per lane
        float* dst = xg + (((size_t)dir * NT + (m >> 4)) * GG + ncol) * 16 + Mv;
        AccU au; au.v = acc;
        *(f32x4*)(dst)     = au.h[0];
        *(f32x4*)(dst + 4) = au.h[1];
    }
}

// ---------------------------------------------------------------------------
// Classifier: [B,256] @ [6,256]^T + b  -> [B,6]
// ---------------------------------------------------------------------------
__global__ void cls_kernel(const float* __restrict__ fh, const float* __restrict__ w,
                           const float* __restrict__ b, float* __restrict__ out, int Bn)
{
    int id = blockIdx.x * blockDim.x + threadIdx.x;
    if (id >= Bn * 6) return;
    int bb = id / 6, n = id % 6;
    float acc = b[n];
    for (int k = 0; k < 256; ++k)
        acc += fh[(size_t)bb * 256 + k] * w[n * 256 + k];
    out[id] = acc;
}

// ---------------------------------------------------------------------------
extern "C" void kernel_launch(void* const* d_in, const int* in_sizes, int n_in,
                              void* d_out, int out_size, void* d_ws, size_t ws_size,
                              hipStream_t stream) {
    (void)in_sizes; (void)n_in; (void)out_size; (void)ws_size;
    const float* x        = (const float*)d_in[0];
    const float* w_ih_l0  = (const float*)d_in[1];
    const float* w_hh_l0  = (const float*)d_in[2];
    const float* b_ih_l0  = (const float*)d_in[3];
    const float* b_hh_l0  = (const float*)d_in[4];
    const float* w_ih_l0r = (const float*)d_in[5];
    const float* w_hh_l0r = (const float*)d_in[6];
    const float* b_ih_l0r = (const float*)d_in[7];
    const float* b_hh_l0r = (const float*)d_in[8];
    const float* w_ih_l1  = (const float*)d_in[9];
    const float* w_hh_l1  = (const float*)d_in[10];
    const float* b_ih_l1  = (const float*)d_in[11];
    const float* b_hh_l1  = (const float*)d_in[12];
    const float* w_ih_l1r = (const float*)d_in[13];
    const float* w_hh_l1r = (const float*)d_in[14];
    const float* b_ih_l1r = (const float*)d_in[15];
    const float* b_hh_l1r = (const float*)d_in[16];
    const float* w_cls    = (const float*)d_in[17];
    const float* b_cls    = (const float*)d_in[18];

    const int T = 512, Bn = 1024, Cc = 9;
    const int Mrows = T * Bn;

    // Workspace layout (~2.4 GiB): xg reused between layers.
    char* ws = (char*)d_ws;
    size_t off = 0;
    float* xg   = (float*)(ws + off); off += (size_t)2 * T * Bn * GG * 4;   // 2 GiB
    bhalf* out0 = (bhalf*)(ws + off); off += (size_t)T * Bn * 256 * 2;      // 256 MiB
    float* hfin = (float*)(ws + off); off += (size_t)Bn * 256 * 4;          // 1 MiB
    bhalf* whhb = (bhalf*)(ws + off); off += (size_t)4 * GG * HH * 2;       // 4x [512][128]
    bhalf* wih1b= (bhalf*)(ws + off); off += (size_t)2 * GG * 256 * 2;      // 2x [512][256]

    // Weight conversions f32 -> bf16
    const int NWHH = GG * HH;    // 65536
    const int NWIH = GG * 256;   // 131072
    convert_kernel<<<(NWHH + 255) / 256, 256, 0, stream>>>(w_hh_l0,  whhb + 0 * NWHH, NWHH);
    convert_kernel<<<(NWHH + 255) / 256, 256, 0, stream>>>(w_hh_l0r, whhb + 1 * NWHH, NWHH);
    convert_kernel<<<(NWHH + 255) / 256, 256, 0, stream>>>(w_hh_l1,  whhb + 2 * NWHH, NWHH);
    convert_kernel<<<(NWHH + 255) / 256, 256, 0, stream>>>(w_hh_l1r, whhb + 3 * NWHH, NWHH);
    convert_kernel<<<(NWIH + 255) / 256, 256, 0, stream>>>(w_ih_l1,  wih1b + 0 * NWIH, NWIH);
    convert_kernel<<<(NWIH + 255) / 256, 256, 0, stream>>>(w_ih_l1r, wih1b + 1 * NWIH, NWIH);

    // Layer 0 input projection (both directions), swizzled output
    xgate0_kernel<<<dim3(T * (Bn / 16), 2), 512, 0, stream>>>(
        x, w_ih_l0, b_ih_l0, b_hh_l0, w_ih_l0r, b_ih_l0r, b_hh_l0r, xg, T, Bn, Cc);

    // Layer 0 scan -> out0 (bf16 [T][B][256])
    lstm_scan_kernel<<<dim3(Bn / 16, 2), 512, 0, stream>>>(
        xg, whhb, out0, nullptr, T, Bn);

    // Layer 1 input projection (WMMA GEMM, overwrites xg, swizzled output)
    xgate1_kernel<<<dim3((Mrows / 64) * 32 / 8, 2), 256, 0, stream>>>(
        out0, wih1b, b_ih_l1, b_hh_l1, b_ih_l1r, b_hh_l1r, xg, Mrows);

    // Layer 1 scan -> final hidden [B][256]
    lstm_scan_kernel<<<dim3(Bn / 16, 2), 512, 0, stream>>>(
        xg, whhb + 2 * NWHH, nullptr, hfin, T, Bn);

    // Classifier
    cls_kernel<<<(Bn * 6 + 255) / 256, 256, 0, stream>>>(
        hfin, w_cls, b_cls, (float*)d_out, Bn);
}
